// CNL_5_85332410237178
// MI455X (gfx1250) — matrix operand order repaired and checked by reading, compiled, not run on hardware
//
#include <hip/hip_runtime.h>

// ---------------------------------------------------------------------------
// CDNA5 (gfx1250) multi-scale non-local attention block.
// All GEMMs run on v_wmma_f32_16x16x32_bf16 (bf16 in, f32 accumulate).
// Softmax cross-lane reductions use DPP16 row_xmask (no LDS round trips).
// ---------------------------------------------------------------------------

typedef unsigned short u16;
typedef __attribute__((ext_vector_type(16))) unsigned short v16u;
typedef __attribute__((ext_vector_type(16))) __bf16        v16bf;
typedef __attribute__((ext_vector_type(8)))  float         v8f;

static __device__ __forceinline__ u16 f2bf(float x) {
    unsigned u = __float_as_uint(x);
    u += 0x7fffu + ((u >> 16) & 1u);          // round-to-nearest-even
    return (u16)(u >> 16);
}

static __device__ __forceinline__ v16bf as_bf(v16u u) {
    return __builtin_bit_cast(v16bf, u);
}

static __device__ __forceinline__ v8f wmma_bf16(v16u a, v16u b, v8f c) {
    // (neg_a, A, neg_b, B, c_mod, C, reuse_a, reuse_b)
    return __builtin_amdgcn_wmma_f32_16x16x32_bf16(
        false, as_bf(a), false, as_bf(b), (short)0, c, false, false);
}

// DPP16 row_xmask:N  (lane <-> lane^N within a 16-lane row; wave32 halves
// never mix for N<16, matching our 16-lane reduction domains).
template <int XMASK>
static __device__ __forceinline__ float dpp_xor(float v) {
    int x = __builtin_amdgcn_mov_dpp(__builtin_bit_cast(int, v),
                                     0x160 + XMASK, 0xf, 0xf, true);
    return __builtin_bit_cast(float, x);
}

static __device__ __forceinline__ float rowmax16(float v) {
    v = fmaxf(v, dpp_xor<1>(v));
    v = fmaxf(v, dpp_xor<2>(v));
    v = fmaxf(v, dpp_xor<4>(v));
    v = fmaxf(v, dpp_xor<8>(v));
    return v;
}

static __device__ __forceinline__ float rowsum16(float v) {
    v += dpp_xor<1>(v);
    v += dpp_xor<2>(v);
    v += dpp_xor<4>(v);
    v += dpp_xor<8>(v);
    return v;
}

// A-matrix 16x32 bf16 tile: lane m=lane&15 holds row m; two contiguous
// 8-element K runs at k0=(lane>>4)*8 and k0+16.  (ISA 7.12.2)
static __device__ __forceinline__ v16u load_a16x32(const u16* base, int ld, int lane) {
    const int m  = lane & 15;
    const int hk = (lane >> 4) * 8;
    const u16* p = base + (long)m * ld + hk;
    union { uint4 q[2]; v16u v; } u;
    u.q[0] = *(const uint4*)(p);
    u.q[1] = *(const uint4*)(p + 16);
    return u.v;
}

// B-matrix 32x16 bf16 tile: lane n=lane&15 holds column n; two contiguous
// 8-element K runs at k0=(lane>>4)*8 and k0+16.  `base` points at (k0, col0) of a
// K-contiguous-per-column (i.e. transposed row-major) buffer.
static __device__ __forceinline__ v16u load_b32x16(const u16* base, int ld, int lane) {
    const int n  = lane & 15;
    const int hk = (lane >> 4) * 8;
    const u16* p = base + (long)n * ld + hk;
    union { uint4 q[2]; v16u v; } u;
    u.q[0] = *(const uint4*)(p);
    u.q[1] = *(const uint4*)(p + 16);
    return u.v;
}

// ---------------------------------------------------------------------------
// Generic batched WMMA GEMM:  D[b] = A[b] (M x K, bf16, K-contig rows)
//                                   @ B[b] (K x N, bf16, K-contig columns)
// One 16x16 D tile per wave.  M,N multiples of 16; K multiple of 32.
// ---------------------------------------------------------------------------
template <int OUT_BF16>
__global__ void cnl5_gemm_k(const u16* __restrict__ A, const u16* __restrict__ B,
                            void* __restrict__ D,
                            int Mt, int Nt, int K, int ldA, int ldB, int ldD,
                            long sA, long sB, long sD) {
    const int lane = threadIdx.x & 31;
    const int wave = threadIdx.x >> 5;
    const int t = blockIdx.x * 8 + wave;
    if (t >= Mt * Nt) return;                     // wave-uniform guard
    const int tm = t / Nt, tn = t % Nt;
    const int bI = blockIdx.y;

    const u16* Ab = A + (long)bI * sA + (long)tm * 16 * ldA;
    const u16* Bb = B + (long)bI * sB + (long)tn * 16 * ldB;

    v8f acc = {};
    for (int k = 0; k < K; k += 32) {
        v16u a = load_a16x32(Ab + k, ldA, lane);
        v16u b = load_b32x16(Bb + k, ldB, lane);
        acc = wmma_bf16(a, b, acc);
    }

    const int col   = tn * 16 + (lane & 15);
    const int rbase = tm * 16 + (lane >> 4) * 8;
    if (OUT_BF16) {
        u16* Dp = (u16*)D + (long)bI * sD;
#pragma unroll
        for (int r = 0; r < 8; ++r) Dp[(long)(rbase + r) * ldD + col] = f2bf(acc[r]);
    } else {
        float* Dp = (float*)D + (long)bI * sD;
#pragma unroll
        for (int r = 0; r < 8; ++r) Dp[(long)(rbase + r) * ldD + col] = acc[r];
    }
}

// ---------------------------------------------------------------------------
// Flash-style attention: one wave owns a 16-row strip of N=2304; online
// softmax over M in 32-column chunks.  TQ[b][N][64] bf16, Pt[b][Mp][64] bf16,
// Gn[b][64][Mp] bf16  ->  Out[b][N][64] f32.
// ---------------------------------------------------------------------------
__global__ void cnl5_attn_k(const u16* __restrict__ TQ, const u16* __restrict__ Pt,
                            const u16* __restrict__ Gn, float* __restrict__ Out,
                            int Mreal, int Mp) {
    __shared__ alignas(16) u16 smp[8][16 * 32];   // prob staging, 1KB/wave

    const int lane = threadIdx.x & 31;
    const int wave = threadIdx.x >> 5;
    const int gw = blockIdx.x * 8 + wave;
    if (gw >= 4 * 144) return;                    // wave-uniform
    const int bI   = gw / 144;
    const int row0 = (gw % 144) * 16;

    const u16* tqb = TQ + (long)bI * (2304 * 64) + (long)row0 * 64;
    const v16u a0 = load_a16x32(tqb, 64, lane);        // K = 0..31  (c)
    const v16u a1 = load_a16x32(tqb + 32, 64, lane);   // K = 32..63
    const u16* ptb = Pt + (long)bI * Mp * 64;
    const u16* gnb = Gn + (long)bI * 64 * Mp;

    v8f o[4] = {};
    float mrun[8], lrun[8];
#pragma unroll
    for (int r = 0; r < 8; ++r) { mrun[r] = -3.0e38f; lrun[r] = 0.0f; }

    const int colm = lane & 15;
    const int half = lane >> 4;
    u16* myp = &smp[wave][0];

    for (int ms = 0; ms < Mp; ms += 32) {
        // ---- logits: two 16x16 tiles, K = 64 (c) ----
        v8f c0 = {}, c1 = {};
        c0 = wmma_bf16(a0, load_b32x16(ptb + (long)ms * 64,           64, lane), c0);
        c0 = wmma_bf16(a1, load_b32x16(ptb + (long)ms * 64 + 32,      64, lane), c0);
        c1 = wmma_bf16(a0, load_b32x16(ptb + (long)(ms + 16) * 64,      64, lane), c1);
        c1 = wmma_bf16(a1, load_b32x16(ptb + (long)(ms + 16) * 64 + 32, 64, lane), c1);
        if (ms + 32 < Mp) {
            __builtin_prefetch(ptb + (long)(ms + 32) * 64, 0, 1);  // global_prefetch_b8
            __builtin_prefetch(gnb + ms + 32, 0, 1);
        }

        const bool v0 = (ms + colm) < Mreal;
        const bool v1 = (ms + 16 + colm) < Mreal;

        // ---- online softmax update (DPP row_xmask butterflies, no LDS) ----
#pragma unroll
        for (int r = 0; r < 8; ++r) {
            float x0 = v0 ? c0[r] : -1.0e30f;
            float x1 = v1 ? c1[r] : -1.0e30f;
            const float tm = rowmax16(fmaxf(x0, x1));
            const float mn = fmaxf(mrun[r], tm);
            const float sc = __expf(mrun[r] - mn);
            const float p0 = __expf(x0 - mn);
            const float p1 = __expf(x1 - mn);
            const float rs = rowsum16(p0 + p1);
            lrun[r] = lrun[r] * sc + rs;
            mrun[r] = mn;
            o[0][r] *= sc; o[1][r] *= sc; o[2][r] *= sc; o[3][r] *= sc;
            const int rr = r + 8 * half;           // C-layout row
            myp[rr * 32 + colm]      = f2bf(p0);
            myp[rr * 32 + 16 + colm] = f2bf(p1);
        }

        // ---- reload probs in A layout, accumulate P @ G ----
        v16u ap;
        {
            const int m = lane & 15, hk = half * 8;
            const u16* p = myp + m * 32 + hk;
            union { uint4 q[2]; v16u v; } u;
            u.q[0] = *(const uint4*)(p);
            u.q[1] = *(const uint4*)(p + 16);
            ap = u.v;
        }
#pragma unroll
        for (int ct = 0; ct < 4; ++ct)
            o[ct] = wmma_bf16(ap, load_b32x16(gnb + (long)(ct * 16) * Mp + ms, Mp, lane), o[ct]);
    }

    float* ob = Out + (long)bI * (2304 * 64);
#pragma unroll
    for (int r = 0; r < 8; ++r) {
        const float inv = 1.0f / lrun[r];
        const int row = row0 + r + 8 * half;
#pragma unroll
        for (int ct = 0; ct < 4; ++ct)
            ob[(long)row * 64 + ct * 16 + colm] = o[ct][r] * inv;
    }
}

// ---------------------------------------------------------------------------
// Transpose + f32->bf16 convert:  in[b][Cr][M] f32 -> out[b][Mp][Cr] bf16,
// zero-padding rows m >= M.
// ---------------------------------------------------------------------------
__global__ void cnl5_tconv_k(const float* __restrict__ in, u16* __restrict__ out,
                             int Bn, int Cr, int M, int Mp) {
    long i = (long)blockIdx.x * blockDim.x + threadIdx.x;
    const long total = (long)Bn * Mp * Cr;
    if (i >= total) return;
    const int c = (int)(i % Cr);
    const long t = i / Cr;
    const int m = (int)(t % Mp);
    const int b = (int)(t / Mp);
    const float v = (m < M) ? in[((long)b * Cr + c) * M + m] : 0.0f;
    out[i] = f2bf(v);
}

__global__ void cnl5_cvt_k(const float* __restrict__ in, u16* __restrict__ out, int n) {
    int i = blockIdx.x * blockDim.x + threadIdx.x;
    if (i < n) out[i] = f2bf(in[i]);
}

// ---------------------------------------------------------------------------
// BatchNorm train-mode stats over (b, h, w) per channel: Z[4][256][2304].
// ---------------------------------------------------------------------------
__global__ void cnl5_bnstats_k(const float* __restrict__ Z, float* __restrict__ stats) {
    __shared__ float ss[256], ss2[256];
    const int cp = blockIdx.x, tid = threadIdx.x;
    float s = 0.0f, s2 = 0.0f;
    for (int i = tid; i < 4 * 2304; i += 256) {
        const int b = i / 2304, sp = i % 2304;
        const float v = Z[((long)b * 256 + cp) * 2304 + sp];
        s += v; s2 += v * v;
    }
    ss[tid] = s; ss2[tid] = s2;
    __syncthreads();
    for (int st = 128; st > 0; st >>= 1) {
        if (tid < st) { ss[tid] += ss[tid + st]; ss2[tid] += ss2[tid + st]; }
        __syncthreads();
    }
    if (tid == 0) {
        const float mean = ss[0] * (1.0f / 9216.0f);
        const float var  = ss2[0] * (1.0f / 9216.0f) - mean * mean;
        stats[2 * cp]     = mean;
        stats[2 * cp + 1] = rsqrtf(var + 1e-5f);
    }
}

__global__ void cnl5_bnfinal_k(const float* __restrict__ Z, const float* __restrict__ stats,
                               const float* __restrict__ gam, const float* __restrict__ bet,
                               float* __restrict__ out, int accum) {
    long i = (long)blockIdx.x * blockDim.x + threadIdx.x;
    if (i >= (long)4 * 256 * 2304) return;
    const int cp = (int)((i / 2304) % 256);
    const float v = (Z[i] - stats[2 * cp]) * stats[2 * cp + 1] * gam[cp] + bet[cp];
    out[i] = accum ? (out[i] + v) : v;
}

// ---------------------------------------------------------------------------
extern "C" void kernel_launch(void* const* d_in, const int* in_sizes, int n_in,
                              void* d_out, int out_size, void* d_ws, size_t ws_size,
                              hipStream_t stream) {
    (void)in_sizes; (void)n_in; (void)out_size; (void)ws_size;

    const float* persp = (const float*)d_in[0];
    const float* resp[5];
    for (int i = 0; i < 5; ++i) resp[i] = (const float*)d_in[1 + i];
    const float* t_w = (const float*)d_in[6];
    const float* z_w = (const float*)d_in[7];
    const float *p_w[5], *g_w[5], *bng[5], *bnb[5];
    for (int i = 0; i < 5; ++i) {
        p_w[i] = (const float*)d_in[8 + 4 * i];
        g_w[i] = (const float*)d_in[9 + 4 * i];
        bng[i] = (const float*)d_in[10 + 4 * i];
        bnb[i] = (const float*)d_in[11 + 4 * i];
    }
    float* out = (float*)d_out;

    const int Cr[5] = {64, 256, 512, 1024, 2048};
    const int Mr[5] = {2304, 2304, 576, 144, 36};
    const int Mp[5] = {2304, 2304, 576, 160, 64};   // padded to multiple of 32

    // ---- workspace carve-out (~27 MB) ----
    char* w = (char*)d_ws;
    auto carve = [&](size_t bytes) -> void* {
        void* p = (void*)w;
        w += (bytes + 255) & ~(size_t)255;
        return p;
    };
    u16*   perspT = (u16*)  carve((size_t)4 * 2304 * 256 * 2);
    u16*   TQ     = (u16*)  carve((size_t)4 * 2304 * 64 * 2);
    u16*   respT  = (u16*)  carve((size_t)4 * 2304 * 256 * 2);  // max Mp*Cr
    u16*   Pt     = (u16*)  carve((size_t)4 * 2304 * 64 * 2);
    u16*   Gn     = (u16*)  carve((size_t)4 * 64 * 2304 * 2);
    float* OutB   = (float*)carve((size_t)4 * 2304 * 64 * 4);
    u16*   Rt     = (u16*)  carve((size_t)4 * 2304 * 64 * 2);
    float* Zb     = (float*)carve((size_t)4 * 256 * 2304 * 4);
    u16*   wT     = (u16*)  carve((size_t)64 * 256 * 2);
    u16*   wZ     = (u16*)  carve((size_t)256 * 64 * 2);
    u16*   wP     = (u16*)  carve((size_t)64 * 2048 * 2);
    u16*   wG     = (u16*)  carve((size_t)64 * 2048 * 2);
    float* stats  = (float*)carve((size_t)512 * 4);

    const int BT = 256;
    auto blocks = [](long n, int bs) { return (int)((n + bs - 1) / bs); };

    // ---- shared stages: weights + perspective transpose + TQ GEMM ----
    cnl5_cvt_k<<<blocks(64 * 256, BT), BT, 0, stream>>>(t_w, wT, 64 * 256);
    cnl5_cvt_k<<<blocks(256 * 64, BT), BT, 0, stream>>>(z_w, wZ, 256 * 64);
    cnl5_tconv_k<<<blocks((long)4 * 2304 * 256, BT), BT, 0, stream>>>(persp, perspT, 4, 256, 2304, 2304);
    // TQ[b][N][64] = perspT[b][N][256] @ t_w^T
    cnl5_gemm_k<1><<<dim3(blocks(144 * 4, 8), 4), BT, 0, stream>>>(
        perspT, wT, TQ, 144, 4, 256, 256, 256, 64,
        (long)2304 * 256, 0, (long)2304 * 64);

    for (int i = 0; i < 5; ++i) {
        const int cr = Cr[i], mr = Mr[i], mp = Mp[i];
        cnl5_cvt_k<<<blocks(64 * cr, BT), BT, 0, stream>>>(p_w[i], wP, 64 * cr);
        cnl5_cvt_k<<<blocks(64 * cr, BT), BT, 0, stream>>>(g_w[i], wG, 64 * cr);
        cnl5_tconv_k<<<blocks((long)4 * mp * cr, BT), BT, 0, stream>>>(resp[i], respT, 4, cr, mr, mp);

        // Pt[b][Mp][64] = respT[b][Mp][Cr] @ p_w^T
        cnl5_gemm_k<1><<<dim3(blocks((mp / 16) * 4, 8), 4), BT, 0, stream>>>(
            respT, wP, Pt, mp / 16, 4, cr, cr, cr, 64,
            (long)mp * cr, 0, (long)mp * 64);
        // Gn[b][64][Mp] = g_w @ resp   (B columns = respT rows)
        cnl5_gemm_k<1><<<dim3(blocks(4 * (mp / 16), 8), 4), BT, 0, stream>>>(
            wG, respT, Gn, 4, mp / 16, cr, cr, cr, mp,
            0, (long)mp * cr, (long)64 * mp);

        // attention -> OutB[b][N][64] f32
        cnl5_attn_k<<<72, BT, 0, stream>>>(TQ, Pt, Gn, OutB, mr, mp);

        // torch-faithful reshape: OutB flat viewed as [b][64][2304]; transpose to Rt[b][s][64]
        cnl5_tconv_k<<<blocks((long)4 * 2304 * 64, BT), BT, 0, stream>>>(OutB, Rt, 4, 64, 2304, 2304);

        // Z[b][256][2304] = z_w @ reshaped
        cnl5_gemm_k<0><<<dim3(blocks(16 * 144, 8), 4), BT, 0, stream>>>(
            wZ, Rt, Zb, 16, 144, 64, 64, 64, 2304,
            0, (long)2304 * 64, (long)256 * 2304);

        cnl5_bnstats_k<<<256, BT, 0, stream>>>(Zb, stats);
        cnl5_bnfinal_k<<<blocks((long)4 * 256 * 2304, BT), BT, 0, stream>>>(
            Zb, stats, bng[i], bnb[i], out, i > 0 ? 1 : 0);
    }
}
